// RankingLoss_14929306321101
// MI455X (gfx1250) — compile-verified, matrix operand order checked by reading, not run on hardware
//
#include <hip/hip_runtime.h>

typedef _Float16 h2   __attribute__((ext_vector_type(2)));
typedef _Float16 h8   __attribute__((ext_vector_type(8)));
typedef _Float16 v16h __attribute__((ext_vector_type(16)));
typedef float    v8f  __attribute__((ext_vector_type(8)));

#define B_DIM   64
#define C_DIM   2048
#define L_DIM   1024
#define NEG_PEN 0.03f
#define BIGV    1024.0f

__global__ void rl_zero_out(float* out) { out[0] = 0.0f; }

// Gather ranks via class_ids and fold the pos/neg masks into the values:
//   rc[b,l] = (label==1) ? r : +BIG   (c-axis, positives)
//   rw[b,l] = (label!=1) ? r : -BIG   (w-axis, negatives)
// so relu(0.03 + rw - rc) == masked penalty with zero masking work later.
__global__ void rl_prep(const float* __restrict__ ranks,
                        const int* __restrict__ labels,
                        const long long* __restrict__ ids,
                        _Float16* __restrict__ rc,
                        _Float16* __restrict__ rw) {
  int i = blockIdx.x * blockDim.x + threadIdx.x;   // over B*L
  if (i >= B_DIM * L_DIM) return;
  int b = i >> 10;                // / L_DIM
  int l = i & (L_DIM - 1);
  int cls = (int)ids[l];
  float r  = ranks[b * C_DIM + cls];
  int  lab = labels[b * C_DIM + cls];
  rc[i] = (lab == 1) ? (_Float16)r : (_Float16)( BIGV);
  rw[i] = (lab != 1) ? (_Float16)r : (_Float16)(-BIGV);
}

// One block = (batch b, 128-row c-tile). Each of 8 waves owns 16 c-rows and
// sweeps all 1024 w in K=32 chunks. Software-pipelined 4 chunks deep: issue
// 8x ds_load_b128, then 64 packed-f16 relu ops + 4 WMMAs. B = ones(32x16)
// turns v_wmma_f32_16x16x32_f16 into a 512-term f32 reduction per issue that
// co-executes with the VALU pen math.
__global__ __launch_bounds__(256) void rl_wmma(const _Float16* __restrict__ rc_g,
                                               const _Float16* __restrict__ rw_g,
                                               float* __restrict__ out) {
  __shared__ __align__(16) h2 srw[L_DIM / 2];   // 2 KB: rw[b,:] as packed pairs
  __shared__ float wsum[8];

  const int b     = blockIdx.x >> 3;
  const int ctile = blockIdx.x & 7;
  const int tid   = threadIdx.x;
  const int wv    = tid >> 5;
  const int lane  = tid & 31;

  const h2* rw2 = (const h2*)(rw_g + b * L_DIM);
  srw[tid]       = rw2[tid];
  srw[tid + 256] = rw2[tid + 256];
  __syncthreads();

  const int c0    = ctile * 128 + wv * 16;
  const float rcv = (float)rc_g[b * L_DIM + c0 + (lane & 15)];
  const _Float16 sh = (_Float16)(NEG_PEN - rcv);   // loop-invariant per lane
  const h2 s2 = { sh, sh };
  const h2 z2 = { (_Float16)0.0f, (_Float16)0.0f };

  v16h ones;
  #pragma unroll
  for (int e = 0; e < 16; ++e) ones[e] = (_Float16)1.0f;

  v8f acc = {0.f, 0.f, 0.f, 0.f, 0.f, 0.f, 0.f, 0.f};

  const int hiK = (lane >= 16) ? 8 : 0;   // K-half per 16-bit A-matrix layout

  for (int w0 = 0; w0 < L_DIM; w0 += 128) {    // 4 K=32 chunks per iteration
    h8 lo[4], hi[4];
    #pragma unroll
    for (int u = 0; u < 4; ++u) {              // batch all LDS loads up front
      const int p = (w0 + u * 32 + hiK) >> 1;  // h2 index, 16B aligned
      lo[u] = *(const h8*)&srw[p];             // k = hiK    .. hiK+7
      hi[u] = *(const h8*)&srw[p + 8];         // k = 16+hiK .. 16+hiK+7
    }
    #pragma unroll
    for (int u = 0; u < 4; ++u) {
      v16h A;
      h2* ap = (h2*)&A;
      const h2* lp = (const h2*)&lo[u];
      const h2* hp = (const h2*)&hi[u];
      #pragma unroll
      for (int q = 0; q < 4; ++q) {            // v_pk_add + v_pk_max per pair
        ap[q]     = __builtin_elementwise_max(lp[q] + s2, z2);
        ap[q + 4] = __builtin_elementwise_max(hp[q] + s2, z2);
      }
      acc = __builtin_amdgcn_wmma_f32_16x16x32_f16(
          false, A, false, ones, (short)0, acc, false, false);
    }
  }

  // Column N=0 lives in lanes 0 and 16; their 8-reg sums give the exact tile total.
  float local = acc[0] + acc[1] + acc[2] + acc[3] +
                acc[4] + acc[5] + acc[6] + acc[7];
  local += __shfl_down(local, 16);
  if (lane == 0) wsum[wv] = local;
  __syncthreads();
  if (tid == 0) {
    float t = 0.f;
    #pragma unroll
    for (int i = 0; i < 8; ++i) t += wsum[i];
    atomicAdd(out, t * (1.0f / B_DIM));
  }
}

extern "C" void kernel_launch(void* const* d_in, const int* in_sizes, int n_in,
                              void* d_out, int out_size, void* d_ws, size_t ws_size,
                              hipStream_t stream) {
  const float*     ranks  = (const float*)d_in[0];
  const int*       labels = (const int*)d_in[1];
  const long long* ids    = (const long long*)d_in[2];
  float* out = (float*)d_out;

  _Float16* rc = (_Float16*)d_ws;            // B*L f16 = 128 KB
  _Float16* rw = rc + B_DIM * L_DIM;         // B*L f16 = 128 KB

  rl_zero_out<<<1, 1, 0, stream>>>(out);
  rl_prep<<<(B_DIM * L_DIM) / 256, 256, 0, stream>>>(ranks, labels, ids, rc, rw);
  rl_wmma<<<B_DIM * 8, 256, 0, stream>>>(rc, rw, out);
}
